// ParallelAttentionBlock_18777597018163
// MI455X (gfx1250) — matrix operand order repaired and checked by reading, not compiled, not run on hardware
//
#include <hip/hip_runtime.h>
#include <hip/hip_bf16.h>

// ---------------------------------------------------------------------------
// Problem constants
// ---------------------------------------------------------------------------
#define B_ 4
#define S_ 2048
#define E_ 1024
#define H_ 16
#define D_ 64
#define M_TOTAL (B_ * S_)          // 8192 rows for all projection GEMMs

typedef __bf16 bf16_t;
typedef __attribute__((ext_vector_type(16))) __bf16 v16bf;
typedef __attribute__((ext_vector_type(8)))  __bf16 v8bf;
typedef __attribute__((ext_vector_type(4)))  __bf16 v4bf;
typedef __attribute__((ext_vector_type(8)))  float  v8f;
typedef __attribute__((ext_vector_type(4)))  unsigned int u32x4;
typedef __attribute__((ext_vector_type(8)))  int i32x8;
typedef __attribute__((ext_vector_type(4)))  int i32x4;

union BFrag { v16bf v; v8bf h[2]; };

// ---------------------------------------------------------------------------
// TDM: issue a 2-D tile load Global->LDS via Tensor Data Mover.
//   tile_d0_dw : tile width in DWORDs (data_size = 4B)
//   tile_d1    : number of rows
//   stride_dw  : row stride in DWORDs
// Descriptor bit layout per CDNA5 ISA ch.8 (Group0 128b, Group1 256b).
// Tensor dims are set equal to tile dims (tile fully in-bounds by
// construction), LDS receives the rows densely packed.
// This toolchain exposes the 6-arg builtin:
//   (u32x4 g0, i32x8 g1, i32x4 g2, i32x4 g3, i32x8 extra, i32 cpol)
// ---------------------------------------------------------------------------
__device__ __forceinline__ void tdm_load_2d(unsigned lds_off, const void* gptr,
                                            unsigned tile_d0_dw, unsigned tile_d1,
                                            unsigned stride_dw) {
  unsigned long long ga = (unsigned long long)gptr;
  u32x4 g0;
  g0[0] = 1u;                                      // count=1 (valid), user mode
  g0[1] = lds_off;                                 // lds_addr (bytes)
  g0[2] = (unsigned)(ga & 0xFFFFFFFFu);            // global_addr[31:0]
  g0[3] = (unsigned)((ga >> 32) & 0x01FFFFFFu)     // global_addr[56:32]
        | 0x80000000u;                             // type = 2 ("image")
  i32x8 g1;
  g1[0] = (int)(2u << 16);                         // data_size=4B, no multicast
  g1[1] = (int)((tile_d0_dw & 0xFFFFu) << 16);     // tensor_dim0[15:0]
  g1[2] = (int)(((tile_d0_dw >> 16) & 0xFFFFu)     // tensor_dim0[31:16]
        |       ((tile_d1 & 0xFFFFu) << 16));      // tensor_dim1[15:0]
  g1[3] = (int)(((tile_d1 >> 16) & 0xFFFFu)        // tensor_dim1[31:16]
        |       ((tile_d0_dw & 0xFFFFu) << 16));   // tile_dim0
  g1[4] = (int)(tile_d1 & 0xFFFFu);                // tile_dim1 (tile_dim2=0)
  g1[5] = (int)stride_dw;                          // tensor_dim0_stride[31:0]
  g1[6] = 0;                                       // stride hi, dim1_stride lo
  g1[7] = 0;
  i32x4 z4 = {0, 0, 0, 0};
  i32x8 z8 = {0, 0, 0, 0, 0, 0, 0, 0};
  __builtin_amdgcn_tensor_load_to_lds(g0, g1, z4, z4, z8, 0);
}

__device__ __forceinline__ unsigned lds_offset_of(const void* p) {
  // Low 32 bits of the shared-aperture flat address are the LDS byte offset.
  return (unsigned)(unsigned long long)p;
}

// ---------------------------------------------------------------------------
// fp32 -> bf16 conversion (vectorized x4)
// ---------------------------------------------------------------------------
__global__ void cvt_f32_bf16(const float* __restrict__ src,
                             bf16_t* __restrict__ dst, int n4) {
  int i = blockIdx.x * blockDim.x + threadIdx.x;
  if (i < n4) {
    float4 f = ((const float4*)src)[i];
    v4bf o;
    o[0] = (bf16_t)f.x; o[1] = (bf16_t)f.y;
    o[2] = (bf16_t)f.z; o[3] = (bf16_t)f.w;
    ((v4bf*)dst)[i] = o;
  }
}

// ---------------------------------------------------------------------------
// Tiled GEMM:  Y[m,n] = sum_k X[m,k] * W[n,k] + bias[n]
//   X: [M,K] bf16 row-major, W: [N,K] bf16 row-major (torch Linear convention)
//   256 threads (8 waves); tile 128(M) x 64(N); K-chunk 32.
//   TDM double-buffered staging: wave 0 posts descriptors for chunk i+1 while
//   all waves run WMMAs on chunk i.
// out_mode: 0 -> bf16 at [B,H,S,D]; 1 -> bf16 at [B,H,D,S]; 2 -> fp32 [M,N]
// ---------------------------------------------------------------------------
__launch_bounds__(256)
__global__ void gemm_xwT(const bf16_t* __restrict__ X,
                         const bf16_t* __restrict__ W,
                         const float*  __restrict__ bias,
                         bf16_t* __restrict__ Ybf,
                         float*  __restrict__ Yf32,
                         int M, int N, int K, int out_mode) {
  __shared__ __align__(16) bf16_t lA[2][128 * 32];   // 2 x 8 KB
  __shared__ __align__(16) bf16_t lB[2][64 * 32];    // 2 x 4 KB

  const int tid  = threadIdx.x;
  const int wave = tid >> 5;
  const int lane = tid & 31;
  const int m0   = blockIdx.x * 128;
  const int n0   = blockIdx.y * 64;

  const unsigned ldsA0 = lds_offset_of(&lA[0][0]), ldsA1 = lds_offset_of(&lA[1][0]);
  const unsigned ldsB0 = lds_offset_of(&lB[0][0]), ldsB1 = lds_offset_of(&lB[1][0]);
  const unsigned strideDW = (unsigned)K >> 1;        // K bf16 = K/2 dwords

  v8f acc[4] = {};

  const int nIter = K / 32;
  if (wave == 0) {   // prime the pipeline: chunk 0 -> buffer 0
    tdm_load_2d(ldsA0, X + (size_t)m0 * K, 16, 128, strideDW);
    tdm_load_2d(ldsB0, W + (size_t)n0 * K, 16, 64,  strideDW);
  }

  for (int i = 0; i < nIter; ++i) {
    const int buf = i & 1;
    if (wave == 0) {
      if (i + 1 < nIter) {
        const int k1 = (i + 1) * 32;
        tdm_load_2d(buf ? ldsA0 : ldsA1, X + (size_t)m0 * K + k1, 16, 128, strideDW);
        tdm_load_2d(buf ? ldsB0 : ldsB1, W + (size_t)n0 * K + k1, 16, 64,  strideDW);
        __builtin_amdgcn_s_wait_tensorcnt(2);   // chunk i done, i+1 in flight
      } else {
        __builtin_amdgcn_s_wait_tensorcnt(0);
      }
    }
    __syncthreads();

    // A fragment (16x32): lane L holds row (L&15); K = {kb..kb+7, kb+16..kb+23}
    BFrag af, bfr[4];
    {
      int arow = wave * 16 + (lane & 15);
      int akb  = (lane >> 4) * 8;
      af.h[0] = *(const v8bf*)&lA[buf][arow * 32 + akb];
      af.h[1] = *(const v8bf*)&lA[buf][arow * 32 + akb + 16];
    }
#pragma unroll
    for (int nb = 0; nb < 4; ++nb) {
      // B fragment (32x16): lane L holds col (L&15); K = {kb..kb+7, kb+16..kb+23}
      int bcol = nb * 16 + (lane & 15);
      int bkb  = (lane >> 4) * 8;
      bfr[nb].h[0] = *(const v8bf*)&lB[buf][bcol * 32 + bkb];
      bfr[nb].h[1] = *(const v8bf*)&lB[buf][bcol * 32 + bkb + 16];
    }
#pragma unroll
    for (int nb = 0; nb < 4; ++nb)
      acc[nb] = __builtin_amdgcn_wmma_f32_16x16x32_bf16(
          false, af.v, false, bfr[nb].v, (short)0, acc[nb], false, false);
    __syncthreads();   // protect buffer 'buf' from the chunk-(i+2) TDM write
  }

  // Epilogue: C/D layout — VGPR r, lane l: M = r + 8*(l>=16), N = l&15
  const int mrow0 = m0 + wave * 16 + (lane >> 4) * 8;
#pragma unroll
  for (int nb = 0; nb < 4; ++nb) {
    const int n  = n0 + nb * 16 + (lane & 15);
    const float bv = bias[n];
#pragma unroll
    for (int r = 0; r < 8; ++r) {
      const int m = mrow0 + r;
      float val = acc[nb][r] + bv;
      if (out_mode == 2) {
        Yf32[(size_t)m * N + n] = val;
      } else {
        int b = m >> 11, s = m & (S_ - 1);
        int h = n >> 6,  d = n & (D_ - 1);
        size_t idx = (out_mode == 0)
          ? (((size_t)(b * H_ + h)) * S_ + s) * D_ + d     // [B,H,S,D]
          : (((size_t)(b * H_ + h)) * D_ + d) * S_ + s;    // [B,H,D,S]
        Ybf[idx] = (bf16_t)val;
      }
    }
  }
}

// ---------------------------------------------------------------------------
// Flash attention: per wave, one 16-row query tile; 32-key chunks streamed
// through LDS by the TDM, double-buffered.
//   Q,K: [B,H,S,D] bf16;  Vt: [B,H,D,S] bf16;  mask: [S,S] int32
//   ctx out: [B,S,E] bf16
// ---------------------------------------------------------------------------
__device__ inline float rowmax16(float v) {
  v = fmaxf(v, __shfl_xor(v, 1, 32));
  v = fmaxf(v, __shfl_xor(v, 2, 32));
  v = fmaxf(v, __shfl_xor(v, 4, 32));
  v = fmaxf(v, __shfl_xor(v, 8, 32));
  return v;
}
__device__ inline float rowsum16(float v) {
  v += __shfl_xor(v, 1, 32);
  v += __shfl_xor(v, 2, 32);
  v += __shfl_xor(v, 4, 32);
  v += __shfl_xor(v, 8, 32);
  return v;
}

__launch_bounds__(256)
__global__ void flash_attn(const bf16_t* __restrict__ Q,
                           const bf16_t* __restrict__ Kc,
                           const bf16_t* __restrict__ Vt,
                           const int*    __restrict__ mask,
                           bf16_t* __restrict__ ctx) {
  __shared__ __align__(16) bf16_t lK[2][32 * 64];     // 2 x 4 KB  [skey][d]
  __shared__ __align__(16) bf16_t lV[2][64 * 32];     // 2 x 4 KB  [d][skey]
  __shared__ __align__(16) bf16_t lP[8][16 * 32];     // 8 KB P relayout scratch

  const int tid  = threadIdx.x;
  const int wave = tid >> 5;
  const int lane = tid & 31;
  const int bh   = blockIdx.y;                   // 0..63
  const int b    = bh >> 4, h = bh & (H_ - 1);
  const int q0   = blockIdx.x * 128 + wave * 16; // this wave's query base

  const bf16_t* Qbh = Q  + (size_t)bh * S_ * D_;
  const bf16_t* Kbh = Kc + (size_t)bh * S_ * D_;
  const bf16_t* Vbh = Vt + (size_t)bh * D_ * S_;

  const unsigned ldsK0 = lds_offset_of(&lK[0][0]), ldsK1 = lds_offset_of(&lK[1][0]);
  const unsigned ldsV0 = lds_offset_of(&lV[0][0]), ldsV1 = lds_offset_of(&lV[1][0]);

  // Preload Q fragments for d-chunks {0..31} and {32..63}
  BFrag qa[2];
  {
    const bf16_t* qr = Qbh + (size_t)(q0 + (lane & 15)) * D_;
    int kb = (lane >> 4) * 8;
    qa[0].h[0] = *(const v8bf*)(qr + kb);
    qa[0].h[1] = *(const v8bf*)(qr + kb + 16);
    qa[1].h[0] = *(const v8bf*)(qr + 32 + kb);
    qa[1].h[1] = *(const v8bf*)(qr + 32 + kb + 16);
  }

  float mrun[8], lrun[8];
  v8f acco[4] = {};
#pragma unroll
  for (int r = 0; r < 8; ++r) { mrun[r] = -1e30f; lrun[r] = 0.f; }

  const float scale = 0.125f;  // 1/sqrt(D), D=64
  const int nChunk = S_ / 32;

  if (wave == 0) {   // prime: chunk 0 -> buffer 0
    tdm_load_2d(ldsK0, Kbh, 1024, 1, 1024);            // 32x64 bf16, contiguous 4KB
    tdm_load_2d(ldsV0, Vbh, 16, 64, S_ / 2);           // 64 rows x 64B, stride 4KB
  }

  for (int i = 0; i < nChunk; ++i) {
    const int buf = i & 1;
    const int s0  = i * 32;
    if (wave == 0) {
      if (i + 1 < nChunk) {
        const int s1 = s0 + 32;
        tdm_load_2d(buf ? ldsK0 : ldsK1, Kbh + (size_t)s1 * D_, 1024, 1, 1024);
        tdm_load_2d(buf ? ldsV0 : ldsV1, Vbh + s1, 16, 64, S_ / 2);
        __builtin_amdgcn_s_wait_tensorcnt(2);
      } else {
        __builtin_amdgcn_s_wait_tensorcnt(0);
      }
    }
    __syncthreads();

    // energy tiles: two 16-key tiles, each = 2 chained wmma over d
    BFrag kf[2][2];
#pragma unroll
    for (int t = 0; t < 2; ++t) {
      int key = t * 16 + (lane & 15);
      int kk  = (lane >> 4) * 8;
      kf[t][0].h[0] = *(const v8bf*)&lK[buf][key * 64 + kk];
      kf[t][0].h[1] = *(const v8bf*)&lK[buf][key * 64 + kk + 16];
      kf[t][1].h[0] = *(const v8bf*)&lK[buf][key * 64 + 32 + kk];
      kf[t][1].h[1] = *(const v8bf*)&lK[buf][key * 64 + 32 + kk + 16];
    }
    v8f et[2];
#pragma unroll
    for (int t = 0; t < 2; ++t) {
      v8f e0 = {};
      e0 = __builtin_amdgcn_wmma_f32_16x16x32_bf16(false, qa[0].v, false, kf[t][0].v,
                                                   (short)0, e0, false, false);
      e0 = __builtin_amdgcn_wmma_f32_16x16x32_bf16(false, qa[1].v, false, kf[t][1].v,
                                                   (short)0, e0, false, false);
      et[t] = e0;
    }

    // scale + mask
#pragma unroll
    for (int t = 0; t < 2; ++t) {
#pragma unroll
      for (int r = 0; r < 8; ++r) {
        int qrow = q0 + r + 8 * (lane >> 4);
        int scol = s0 + t * 16 + (lane & 15);
        float e = et[t][r] * scale;
        if (mask[(size_t)qrow * S_ + scol] == 0) e = -1e9f;
        et[t][r] = e;
      }
    }

    // online softmax update + P relayout into lP (C layout -> A layout via LDS)
#pragma unroll
    for (int r = 0; r < 8; ++r) {
      float cm = rowmax16(fmaxf(et[0][r], et[1][r]));
      float nm = fmaxf(mrun[r], cm);
      float corr = __expf(mrun[r] - nm);
      mrun[r] = nm;
      float p0 = __expf(et[0][r] - nm);
      float p1 = __expf(et[1][r] - nm);
      lrun[r] = lrun[r] * corr + rowsum16(p0 + p1);
#pragma unroll
      for (int nb = 0; nb < 4; ++nb) acco[nb][r] *= corr;
      int prow = r + 8 * (lane >> 4);
      lP[wave][prow * 32 + (lane & 15)]      = (bf16_t)p0;
      lP[wave][prow * 32 + 16 + (lane & 15)] = (bf16_t)p1;
    }

    // context: acco[m,d] += P(16x32) x V(32x16 per d-subtile)
    BFrag pa, vb[4];
    {
      int prow = lane & 15;
      int pkb  = (lane >> 4) * 8;
      pa.h[0] = *(const v8bf*)&lP[wave][prow * 32 + pkb];
      pa.h[1] = *(const v8bf*)&lP[wave][prow * 32 + pkb + 16];
    }
#pragma unroll
    for (int nb = 0; nb < 4; ++nb) {
      int dcol = nb * 16 + (lane & 15);
      int sk   = (lane >> 4) * 8;
      vb[nb].h[0] = *(const v8bf*)&lV[buf][dcol * 32 + sk];
      vb[nb].h[1] = *(const v8bf*)&lV[buf][dcol * 32 + sk + 16];
    }
#pragma unroll
    for (int nb = 0; nb < 4; ++nb)
      acco[nb] = __builtin_amdgcn_wmma_f32_16x16x32_bf16(
          false, pa.v, false, vb[nb].v, (short)0, acco[nb], false, false);
    __syncthreads();
  }

  // finalize: ctx[b, q, h*64+d] = acco / lrun
#pragma unroll
  for (int nb = 0; nb < 4; ++nb) {
    int d = nb * 16 + (lane & 15);
#pragma unroll
    for (int r = 0; r < 8; ++r) {
      int q = q0 + r + 8 * (lane >> 4);
      float val = acco[nb][r] / fmaxf(lrun[r], 1e-20f);
      ctx[((size_t)b * S_ + q) * E_ + h * D_ + d] = (bf16_t)val;
    }
  }
}

// ---------------------------------------------------------------------------
// Launch: cvt -> 3 projection GEMMs -> flash attention -> output GEMM
// ---------------------------------------------------------------------------
extern "C" void kernel_launch(void* const* d_in, const int* in_sizes, int n_in,
                              void* d_out, int out_size, void* d_ws, size_t ws_size,
                              hipStream_t stream) {
  (void)in_sizes; (void)n_in; (void)out_size; (void)ws_size;

  const float* q_vec = (const float*)d_in[0];
  const float* k_vec = (const float*)d_in[1];
  const float* v_vec = (const float*)d_in[2];
  const int*   mask  = (const int*)  d_in[3];
  const float* Wq = (const float*)d_in[4];  const float* bq = (const float*)d_in[5];
  const float* Wk = (const float*)d_in[6];  const float* bk = (const float*)d_in[7];
  const float* Wv = (const float*)d_in[8];  const float* bv = (const float*)d_in[9];
  const float* Wo = (const float*)d_in[10]; const float* bo = (const float*)d_in[11];
  float* out = (float*)d_out;

  const size_t ME = (size_t)M_TOTAL * E_;   // 8M elems
  const size_t EE = (size_t)E_ * E_;        // 1M elems
  bf16_t* ws  = (bf16_t*)d_ws;
  bf16_t* qx  = ws;
  bf16_t* kx  = qx + ME;
  bf16_t* vx  = kx + ME;
  bf16_t* wq  = vx + ME;
  bf16_t* wk  = wq + EE;
  bf16_t* wv  = wk + EE;
  bf16_t* wo  = wv + EE;
  bf16_t* Qh  = wo + EE;                    // [B,H,S,D]
  bf16_t* Kh  = Qh + ME;                    // [B,H,S,D]
  bf16_t* Vt  = Kh + ME;                    // [B,H,D,S]
  bf16_t* ctx = Vt + ME;                    // [B,S,E]

  auto cvt = [&](const float* s, bf16_t* d, size_t n) {
    int n4 = (int)(n / 4);
    cvt_f32_bf16<<<(n4 + 255) / 256, 256, 0, stream>>>(s, d, n4);
  };
  cvt(q_vec, qx, ME); cvt(k_vec, kx, ME); cvt(v_vec, vx, ME);
  cvt(Wq, wq, EE); cvt(Wk, wk, EE); cvt(Wv, wv, EE); cvt(Wo, wo, EE);

  dim3 gg(M_TOTAL / 128, E_ / 64), gb(256);
  gemm_xwT<<<gg, gb, 0, stream>>>(qx, wq, bq, Qh, nullptr, M_TOTAL, E_, E_, 0);
  gemm_xwT<<<gg, gb, 0, stream>>>(kx, wk, bk, Kh, nullptr, M_TOTAL, E_, E_, 0);
  gemm_xwT<<<gg, gb, 0, stream>>>(vx, wv, bv, Vt, nullptr, M_TOTAL, E_, E_, 1);

  flash_attn<<<dim3(S_ / 128, B_ * H_), 256, 0, stream>>>(Qh, Kh, Vt, mask, ctx);

  gemm_xwT<<<gg, gb, 0, stream>>>(ctx, wo, bo, nullptr, out, M_TOTAL, E_, E_, 2);
}
